// MultiHeadAttention_57312043598036
// MI455X (gfx1250) — matrix operand order repaired and checked by reading, not compiled, not run on hardware
//
#include <hip/hip_runtime.h>
#include <hip/hip_bf16.h>
#include <cmath>

// ---------------------------------------------------------------------------
// MultiHeadAttention for MI455X (gfx1250): all matmuls via v_wmma_f32_16x16x32_bf16
//   1) fp32 -> bf16 conversion of Q,K,V,Wo ; transposed conversion of Wq/Wk/Wv
//   2) projection GEMMs (register double-buffered; v written transposed)
//   3) flash-attention per 16-row query tile (WMMA scores + LDS softmax relayout)
//   4) output projection GEMM: async-LDS-staged B tiles (ASYNCcnt path) + WMMA
// ---------------------------------------------------------------------------

typedef __bf16 bf16_t;
typedef __attribute__((ext_vector_type(16))) __bf16 v16bf;
typedef __attribute__((ext_vector_type(8)))  float  v8f;
typedef __attribute__((ext_vector_type(4)))  int    v4i;

#define DEVFN __device__ __forceinline__

namespace {
constexpr int BATCH = 2;
constexpr int SEQ   = 2048;
constexpr int DM    = 1024;
constexpr int NH    = 16;
constexpr int HD    = 64;   // d_k
}

#if __has_builtin(__builtin_amdgcn_global_load_async_to_lds_b128)
#define MHA_HAVE_ASYNC_LDS 1
#else
#define MHA_HAVE_ASYNC_LDS 0
#endif

DEVFN v8f wmma_bf16(v16bf a, v16bf b, v8f c) {
  // D = A(16x32) * B(32x16) + C(16x16), f32 accumulate
  return __builtin_amdgcn_wmma_f32_16x16x32_bf16(
      /*neg_a=*/false, a, /*neg_b=*/false, b,
      /*c_mod=*/(short)0, c, /*reuse_a=*/false, /*reuse_b=*/false);
}

// A-matrix (16x32 bf16) loader. Lane holds row m=lane&15; lanes 0-15: K in
// [0..7] (v0..3) and [16..23] (v4..7); lanes 16-31: +8.  => two contiguous
// 16B runs per lane at row offsets 8*half and 16+8*half.
DEVFN v16bf load_a16(const bf16_t* base, int lda, int lane) {
  int m = lane & 15, hv = lane >> 4;
  const bf16_t* p = base + (size_t)m * lda + 8 * hv;
  union { v16bf v; uint4 q[2]; } u;
  u.q[0] = *(const uint4*)(p);
  u.q[1] = *(const uint4*)(p + 16);
  return u.v;
}

// B-matrix (32x16 bf16) loader. Lane holds column n=lane&15; lanes 0-15 hold
// K=[0..7] and [16..23], lanes 16-31: +8 => two 16B runs per lane (same as A).
DEVFN v16bf load_b16(const bf16_t* colbase, int ldb, int lane) {
  int n = lane & 15, hv = lane >> 4;
  const bf16_t* p = colbase + (size_t)n * ldb + 8 * hv;
  union { v16bf v; uint4 q[2]; } u;
  u.q[0] = *(const uint4*)(p);
  u.q[1] = *(const uint4*)(p + 16);
  return u.v;
}

#if MHA_HAVE_ASYNC_LDS
// 16B global -> LDS async copy (tracked by ASYNCcnt, overlaps with compute)
DEVFN void async_copy16(const void* g, void* l) {
  __builtin_amdgcn_global_load_async_to_lds_b128(
      (__attribute__((address_space(1))) v4i*)g,
      (__attribute__((address_space(3))) v4i*)l,
      /*imm offset=*/0, /*cpol=*/0);
}
#endif

// ------------------------- conversion kernels ------------------------------

__global__ void mha_cvt_f32_bf16(const float* __restrict__ src,
                                 bf16_t* __restrict__ dst, int n) {
  int i = blockIdx.x * 256 + threadIdx.x;
  if (i < n) dst[i] = (bf16_t)src[i];
}

// W[h][d][k] (f32) -> WT[h][k][d] (bf16) so B-matrix columns are contiguous.
__global__ void mha_cvt_transpose_w(const float* __restrict__ W,
                                    bf16_t* __restrict__ WT, int n) {
  int i = blockIdx.x * 256 + threadIdx.x;
  if (i >= n) return;
  int h = i / (HD * DM);
  int r = i - h * (HD * DM);
  int k = r / DM;
  int d = r - k * DM;
  WT[i] = (bf16_t)W[((size_t)h * DM + d) * HD + k];
}

// ------------------------- projection GEMM ---------------------------------
// per (b,h): out(2048x64) = X_b(2048x1024) @ W_h(1024x64) + bias_h
// grid = (SEQ/64, BATCH*NH), block = 128 (4 waves, each wave 16 rows x 64 cols)
// Register double-buffered: loads for k+1 issued before WMMAs for k.
__global__ void mha_proj_gemm(const bf16_t* __restrict__ Xb,   // [B][SEQ][DM]
                              const bf16_t* __restrict__ WT,   // [NH][HD][DM]
                              const float*  __restrict__ bias, // [NH][HD]
                              bf16_t* __restrict__ out,
                              int transpose_out) {
  int wave = threadIdx.x >> 5, lane = threadIdx.x & 31;
  int b = blockIdx.y >> 4, h = blockIdx.y & 15;
  int m0 = blockIdx.x * 64 + wave * 16;

  const bf16_t* A  = Xb + ((size_t)b * SEQ + m0) * DM;
  const bf16_t* Wh = WT + (size_t)h * HD * DM;

  v8f acc[4];
#pragma unroll
  for (int t = 0; t < 4; ++t)
#pragma unroll
    for (int v = 0; v < 8; ++v) acc[t][v] = 0.0f;

  v16bf a = load_a16(A, DM, lane);
  v16bf bm[4];
#pragma unroll
  for (int t = 0; t < 4; ++t) bm[t] = load_b16(Wh + (size_t)(t * 16) * DM, DM, lane);

#pragma unroll 2
  for (int k0 = 0; k0 < DM; k0 += 32) {
    int kn = (k0 + 32 < DM) ? (k0 + 32) : k0;  // uniform; last iter reloads (harmless)
    __builtin_prefetch((const void*)(A + kn + 128), 0, 1);
    v16bf an = load_a16(A + kn, DM, lane);
    v16bf bn[4];
#pragma unroll
    for (int t = 0; t < 4; ++t)
      bn[t] = load_b16(Wh + (size_t)(t * 16) * DM + kn, DM, lane);
#pragma unroll
    for (int t = 0; t < 4; ++t) acc[t] = wmma_bf16(a, bm[t], acc[t]);
    a = an;
#pragma unroll
    for (int t = 0; t < 4; ++t) bm[t] = bn[t];
  }

  int nlo = lane & 15, hv = lane >> 4;
#pragma unroll
  for (int t = 0; t < 4; ++t) {
    int n = t * 16 + nlo;
    float bv = bias[h * HD + n];
#pragma unroll
    for (int v = 0; v < 8; ++v) {
      int m = m0 + v + 8 * hv;
      float val = acc[t][v] + bv;
      if (!transpose_out)  // [B][H][SEQ][HD]
        out[(((size_t)b * NH + h) * SEQ + m) * HD + n] = (bf16_t)val;
      else                 // [B][H][HD][SEQ]  (v transposed for P@V B-layout)
        out[(((size_t)b * NH + h) * HD + n) * SEQ + m] = (bf16_t)val;
    }
  }
}

// ------------------------- flash attention ---------------------------------
// grid = (SEQ/64, BATCH*NH), block = 128. Each wave: 16 query rows.
__global__ void mha_attn(const bf16_t* __restrict__ qp,  // [B][H][SEQ][HD]
                         const bf16_t* __restrict__ kp,  // [B][H][SEQ][HD]
                         const bf16_t* __restrict__ vT,  // [B][H][HD][SEQ]
                         bf16_t* __restrict__ conc) {    // [B][SEQ][DM]
  __shared__ __align__(16) float  sS[4][16][32];   // raw scores
  __shared__ __align__(16) bf16_t sP[4][16][32];   // exp(P) as A-matrix source
  __shared__ __align__(16) float  sScale[4][16];   // per-row rescale alpha
  __shared__ __align__(16) float  sInvL[4][16];    // per-row 1/l

  int wave = threadIdx.x >> 5, lane = threadIdx.x & 31;
  int b = blockIdx.y >> 4, h = blockIdx.y & 15;
  int m0 = blockIdx.x * 64 + wave * 16;
  int nlo = lane & 15, hv = lane >> 4;

  const bf16_t* Qh = qp + ((size_t)(b * NH + h) * SEQ + m0) * HD;
  const bf16_t* Kh = kp + (size_t)(b * NH + h) * SEQ * HD;
  const bf16_t* Vh = vT + (size_t)(b * NH + h) * HD * SEQ;

  const float scl = 0.125f;  // 1/sqrt(64)
  float mrow = -INFINITY, lrow = 0.0f;  // owned by lanes 0..15 (row = lane)

  v8f o[4];
#pragma unroll
  for (int t = 0; t < 4; ++t)
#pragma unroll
    for (int v = 0; v < 8; ++v) o[t][v] = 0.0f;

  for (int kb = 0; kb < SEQ; kb += 32) {
    // ---- scores: 16x32 tile = two 16x16 WMMA outputs, K-dim = HD (2 steps)
    v8f s0, s1;
#pragma unroll
    for (int v = 0; v < 8; ++v) { s0[v] = 0.0f; s1[v] = 0.0f; }
#pragma unroll
    for (int kk = 0; kk < HD; kk += 32) {
      v16bf a  = load_a16(Qh + kk, HD, lane);
      v16bf b0 = load_b16(Kh + (size_t)kb * HD + kk, HD, lane);
      v16bf b1 = load_b16(Kh + (size_t)(kb + 16) * HD + kk, HD, lane);
      s0 = wmma_bf16(a, b0, s0);
      s1 = wmma_bf16(a, b1, s1);
    }
    // D-layout -> LDS row-major
#pragma unroll
    for (int v = 0; v < 8; ++v) {
      sS[wave][v + 8 * hv][nlo]      = s0[v];
      sS[wave][v + 8 * hv][nlo + 16] = s1[v];
    }
    __syncthreads();

    // ---- online softmax: lanes 0..15 each own one row
    if (lane < 16) {
      float sv[32];
      float bmax = -INFINITY;
#pragma unroll
      for (int j = 0; j < 32; ++j) {
        sv[j] = sS[wave][lane][j] * scl;
        bmax = fmaxf(bmax, sv[j]);
      }
      float mnew  = fmaxf(mrow, bmax);
      float alpha = __expf(mrow - mnew);
      float psum  = 0.0f;
#pragma unroll
      for (int j = 0; j < 32; ++j) {
        float p = __expf(sv[j] - mnew);
        psum += p;
        sP[wave][lane][j] = (bf16_t)p;
      }
      lrow = lrow * alpha + psum;
      mrow = mnew;
      sScale[wave][lane] = alpha;
    }
    __syncthreads();

    // ---- rescale running output (acc elem v holds row v+8*half)
#pragma unroll
    for (int v = 0; v < 8; ++v) {
      float al = sScale[wave][v + 8 * hv];
#pragma unroll
      for (int t = 0; t < 4; ++t) o[t][v] *= al;
    }

    // ---- O += P(16x32) @ V(32x64); P from LDS as A, V columns from vT rows
    v16bf pa = load_a16(&sP[wave][0][0], 32, lane);
#pragma unroll
    for (int t = 0; t < 4; ++t) {
      v16bf bv = load_b16(Vh + (size_t)(t * 16) * SEQ + kb, SEQ, lane);
      o[t] = wmma_bf16(pa, bv, o[t]);
    }
    __syncthreads();  // before next iteration overwrites sS/sP
  }

  if (lane < 16) sInvL[wave][lane] = 1.0f / lrow;
  __syncthreads();

  // ---- normalize + head-order concat store: conc[b][s][h*64 + d]
#pragma unroll
  for (int t = 0; t < 4; ++t) {
    int d = t * 16 + nlo;
#pragma unroll
    for (int v = 0; v < 8; ++v) {
      int m = m0 + v + 8 * hv;
      float val = o[t][v] * sInvL[wave][v + 8 * hv];
      conc[((size_t)b * SEQ + m) * DM + h * HD + d] = (bf16_t)val;
    }
  }
}

// ------------------------- output projection -------------------------------
// out(4096x1024) = concat(4096x1024) @ Wo^T + bo  (Wo row-major => B cols contiguous)
// grid = (M/64, N/64), block = 128. All 4 waves share the same B tile per k-step,
// so B is staged in LDS with double-buffered async copies (ASYNCcnt path).
__global__ void mha_out_gemm(const bf16_t* __restrict__ Cc,   // [B*SEQ][DM]
                             const bf16_t* __restrict__ Wob,  // [DM][DM] row-major
                             const float*  __restrict__ bo,   // [DM]
                             float* __restrict__ out) {       // [B*SEQ][DM] f32
  int tid = threadIdx.x;
  int wave = tid >> 5, lane = tid & 31;
  int m0 = blockIdx.x * 64 + wave * 16;
  int n0 = blockIdx.y * 64;

  const bf16_t* A = Cc + (size_t)m0 * DM;

  v8f acc[4];
#pragma unroll
  for (int t = 0; t < 4; ++t)
#pragma unroll
    for (int v = 0; v < 8; ++v) acc[t][v] = 0.0f;

#if MHA_HAVE_ASYNC_LDS
  // sB[buf][col 0..63][k 0..31]: exact copy of global B-column layout.
  __shared__ __align__(16) bf16_t sB[2][64][32];  // 8 KiB

  // 256 16B-chunks per k-step tile; each of 128 threads copies 2 chunks.
  auto issueB = [&](int buf, int k0) {
#pragma unroll
    for (int i = 0; i < 2; ++i) {
      int j = tid * 2 + i;
      int col = j >> 2, part = j & 3;
      const bf16_t* g = Wob + (size_t)(n0 + col) * DM + k0 + part * 8;
      async_copy16((const void*)g, (void*)&sB[buf][col][part * 8]);
    }
  };

  issueB(0, 0);
  v16bf a = load_a16(A, DM, lane);

  for (int k0 = 0; k0 < DM; k0 += 32) {
    int buf = (k0 >> 5) & 1;
    // current buffer's async copies complete (per-wave), then block-wide sync
    asm volatile("s_wait_asynccnt 0x0" ::: "memory");
    __syncthreads();
    if (k0 + 32 < DM) issueB(buf ^ 1, k0 + 32);  // overlap next tile with compute

    int kn = (k0 + 32 < DM) ? (k0 + 32) : k0;
    v16bf an = load_a16(A + kn, DM, lane);       // double-buffered A
#pragma unroll
    for (int t = 0; t < 4; ++t) {
      v16bf bm = load_b16(&sB[buf][t * 16][0], 32, lane);  // ds_load_b128
      acc[t] = wmma_bf16(a, bm, acc[t]);
    }
    a = an;
    // No trailing barrier needed: a wave only reaches the next iteration's
    // barrier after the s_wait_dscnt feeding this iteration's WMMAs, so the
    // buffer overwritten in the next iteration has no reads in flight.
  }
#else
  // Fallback: direct global B loads, register double-buffered.
  v16bf a = load_a16(A, DM, lane);
  v16bf bm[4];
#pragma unroll
  for (int t = 0; t < 4; ++t)
    bm[t] = load_b16(Wob + (size_t)(n0 + t * 16) * DM, DM, lane);

#pragma unroll 2
  for (int k0 = 0; k0 < DM; k0 += 32) {
    int kn = (k0 + 32 < DM) ? (k0 + 32) : k0;
    __builtin_prefetch((const void*)(A + kn + 128), 0, 1);
    v16bf an = load_a16(A + kn, DM, lane);
    v16bf bn[4];
#pragma unroll
    for (int t = 0; t < 4; ++t)
      bn[t] = load_b16(Wob + (size_t)(n0 + t * 16) * DM + kn, DM, lane);
#pragma unroll
    for (int t = 0; t < 4; ++t) acc[t] = wmma_bf16(a, bm[t], acc[t]);
    a = an;
#pragma unroll
    for (int t = 0; t < 4; ++t) bm[t] = bn[t];
  }
#endif

  int nlo = lane & 15, hv = lane >> 4;
#pragma unroll
  for (int t = 0; t < 4; ++t) {
    int n = n0 + t * 16 + nlo;
    float bv = bo[n];
#pragma unroll
    for (int v = 0; v < 8; ++v) {
      int m = m0 + v + 8 * hv;
      out[(size_t)m * DM + n] = acc[t][v] + bv;
    }
  }
}

// ---------------------------------------------------------------------------

extern "C" void kernel_launch(void* const* d_in, const int* in_sizes, int n_in,
                              void* d_out, int out_size, void* d_ws, size_t ws_size,
                              hipStream_t stream) {
  (void)in_sizes; (void)n_in; (void)out_size; (void)ws_size;

  const float* Q  = (const float*)d_in[0];
  const float* K  = (const float*)d_in[1];
  const float* V  = (const float*)d_in[2];
  const float* Wq = (const float*)d_in[3];
  const float* bq = (const float*)d_in[4];
  const float* Wk = (const float*)d_in[5];
  const float* bk = (const float*)d_in[6];
  const float* Wv = (const float*)d_in[7];
  const float* bv = (const float*)d_in[8];
  const float* Wo = (const float*)d_in[9];
  const float* bo = (const float*)d_in[10];
  float* out = (float*)d_out;

  char* ws = (char*)d_ws;
  const size_t MB = 1024u * 1024u;
  bf16_t* Qb   = (bf16_t*)(ws + 0);        // 8 MiB  [B][SEQ][DM]
  bf16_t* Kb   = (bf16_t*)(ws + 8  * MB);  // 8 MiB
  bf16_t* Vb   = (bf16_t*)(ws + 16 * MB);  // 8 MiB
  bf16_t* WqT  = (bf16_t*)(ws + 24 * MB);  // 2 MiB  [H][HD][DM]
  bf16_t* WkT  = (bf16_t*)(ws + 26 * MB);  // 2 MiB
  bf16_t* WvT  = (bf16_t*)(ws + 28 * MB);  // 2 MiB
  bf16_t* Wob  = (bf16_t*)(ws + 30 * MB);  // 2 MiB  [DM][DM]
  bf16_t* qp   = (bf16_t*)(ws + 32 * MB);  // 8 MiB  [B][H][SEQ][HD]
  bf16_t* kp   = (bf16_t*)(ws + 40 * MB);  // 8 MiB
  bf16_t* vTp  = (bf16_t*)(ws + 48 * MB);  // 8 MiB  [B][H][HD][SEQ]
  bf16_t* conc = (bf16_t*)(ws + 0);        // reuse Qb region (dead after proj)

  const int nQ  = BATCH * SEQ * DM;   // 4,194,304
  const int nW  = NH * HD * DM;       // 1,048,576
  const int nWo = DM * DM;            // 1,048,576

  mha_cvt_f32_bf16<<<(nQ + 255) / 256, 256, 0, stream>>>(Q, Qb, nQ);
  mha_cvt_f32_bf16<<<(nQ + 255) / 256, 256, 0, stream>>>(K, Kb, nQ);
  mha_cvt_f32_bf16<<<(nQ + 255) / 256, 256, 0, stream>>>(V, Vb, nQ);
  mha_cvt_f32_bf16<<<(nWo + 255) / 256, 256, 0, stream>>>(Wo, Wob, nWo);
  mha_cvt_transpose_w<<<(nW + 255) / 256, 256, 0, stream>>>(Wq, WqT, nW);
  mha_cvt_transpose_w<<<(nW + 255) / 256, 256, 0, stream>>>(Wk, WkT, nW);
  mha_cvt_transpose_w<<<(nW + 255) / 256, 256, 0, stream>>>(Wv, WvT, nW);

  dim3 gProj(SEQ / 64, BATCH * NH);
  mha_proj_gemm<<<gProj, 128, 0, stream>>>(Qb, WqT, bq, qp, 0);
  mha_proj_gemm<<<gProj, 128, 0, stream>>>(Kb, WkT, bk, kp, 0);
  mha_proj_gemm<<<gProj, 128, 0, stream>>>(Vb, WvT, bv, vTp, 1);

  mha_attn<<<dim3(SEQ / 64, BATCH * NH), 128, 0, stream>>>(qp, kp, vTp, conc);

  mha_out_gemm<<<dim3(BATCH * SEQ / 64, DM / 64), 128, 0, stream>>>(conc, Wob, bo, out);
}
